// GCNPolicy_55765855371830
// MI455X (gfx1250) — matrix operand order repaired and checked by reading, not compiled, not run on hardware
//
#include <hip/hip_runtime.h>
#include <math.h>

// ---------------- WMMA types (gfx1250, wave32) ----------------
typedef __attribute__((ext_vector_type(16))) __bf16 v16bf;
typedef __attribute__((ext_vector_type(8)))  float  v8f;
typedef __attribute__((ext_vector_type(8)))  unsigned short u16x8;
typedef __attribute__((ext_vector_type(4)))  unsigned short u16x4;

union Frag16 {
    v16bf v;
    u16x8 h2[2];
    unsigned short u[16];
};

__device__ __forceinline__ unsigned short f2bf(float f) {
    unsigned int x = __float_as_uint(f);
    x += 0x7fffu + ((x >> 16) & 1u);          // round-to-nearest-even
    return (unsigned short)(x >> 16);
}

// B-matrix LDS swizzle: fragment-contiguous per lane.
// 16-bit B 32x16 layout: lane = h*16 + (col&15); i<8 -> k = kt*32 + h*8 + i; i>=8 -> +16
__device__ __forceinline__ int bswz(int k, int col) {
    int kt = k >> 5, kk = k & 31;
    int h  = (kk >> 3) & 1, i = ((kk >> 4) << 3) | (kk & 7);
    int nt = col >> 4;
    int lane = (h << 4) | (col & 15);
    return (((kt << 3) + nt) << 9) + (lane << 4) + i;   // ((kt*8+nt)*32+lane)*16 + i
}

// A-matrix LDS swizzle for the edge kernel.
// 16-bit A 16x32 layout: lane = half*16 + m; i<8 -> k = kt*32 + half*8 + i; i>=8 -> +16
__device__ __forceinline__ int aswz(int m, int k) {
    int kt = k >> 5, kk = k & 31;
    int g = kk >> 4, rem = kk & 15;
    int half = rem >> 3;
    int i = (g << 3) | (rem & 7);
    int lane = (half << 4) | m;
    return (kt << 9) + (lane << 4) + i;                  // (kt*32+lane)*16 + i
}

// ---------------- weight convert + zero-pad to Kpad ----------------
__global__ void cvt_pad_bf16(const float* __restrict__ src, unsigned short* __restrict__ dst,
                             int K, int Kpad) {
    int i = blockIdx.x * blockDim.x + threadIdx.x;     // over Kpad*128
    if (i >= Kpad * 128) return;
    int k = i >> 7, c = i & 127;
    dst[i] = (k < K) ? f2bf(src[k * 128 + c]) : (unsigned short)0;
}

__global__ void zero_f32(float* p, int n) {
    int i = blockIdx.x * blockDim.x + threadIdx.x;
    if (i < n) p[i] = 0.f;
}

// ---------------- generic GEMM: out[M,128] = act(A[M,K] @ Bw[Kpad,128] + bias) ----------------
// 4 waves/block, 32 rows per wave (2 M-subtiles -> 2 WMMAs per B-fragment), N=128.
template <int KT>
__global__ void __launch_bounds__(128)
gemm128(const float* __restrict__ A, const unsigned short* __restrict__ Bw,
        const float* __restrict__ bias, const float* __restrict__ shift,
        const float* __restrict__ scale, float* __restrict__ out,
        int M, int K, int relu) {
    extern __shared__ unsigned short sB[];               // KT*32*128 ushorts, swizzled
    const int tid = threadIdx.x, lane = tid & 31, wave = tid >> 5;

    for (int i = tid; i < KT * 32 * 128; i += 128) {
        int k = i >> 7, c = i & 127;
        sB[bswz(k, c)] = Bw[i];
    }
    __syncthreads();

    const int rowBase = blockIdx.x * 128 + wave * 32;
    const int half = lane >> 4;
    const bool fastK = (K == KT * 32) && (shift == nullptr);

    Frag16 af[2][KT];
#pragma unroll
    for (int s = 0; s < 2; ++s) {
        const int row = rowBase + s * 16 + (lane & 15);
        const bool rvalid = row < M;
        const float* arow = A + (long long)row * K;
#pragma unroll
        for (int kt = 0; kt < KT; ++kt) {
            const int k0 = kt * 32 + half * 8;
            if (fastK) {
                if (rvalid) {
                    float4 a0 = *(const float4*)(arow + k0);
                    float4 a1 = *(const float4*)(arow + k0 + 4);
                    float4 b0 = *(const float4*)(arow + k0 + 16);
                    float4 b1 = *(const float4*)(arow + k0 + 20);
                    af[s][kt].u[0]  = f2bf(a0.x); af[s][kt].u[1]  = f2bf(a0.y);
                    af[s][kt].u[2]  = f2bf(a0.z); af[s][kt].u[3]  = f2bf(a0.w);
                    af[s][kt].u[4]  = f2bf(a1.x); af[s][kt].u[5]  = f2bf(a1.y);
                    af[s][kt].u[6]  = f2bf(a1.z); af[s][kt].u[7]  = f2bf(a1.w);
                    af[s][kt].u[8]  = f2bf(b0.x); af[s][kt].u[9]  = f2bf(b0.y);
                    af[s][kt].u[10] = f2bf(b0.z); af[s][kt].u[11] = f2bf(b0.w);
                    af[s][kt].u[12] = f2bf(b1.x); af[s][kt].u[13] = f2bf(b1.y);
                    af[s][kt].u[14] = f2bf(b1.z); af[s][kt].u[15] = f2bf(b1.w);
                } else {
#pragma unroll
                    for (int i = 0; i < 16; ++i) af[s][kt].u[i] = 0;
                }
            } else {
#pragma unroll
                for (int i = 0; i < 8; ++i) {
                    int ka = k0 + i, kb = k0 + 16 + i;
                    float va = 0.f, vb = 0.f;
                    if (rvalid && ka < K) { va = arow[ka]; if (shift) va = (va + shift[ka]) * scale[ka]; }
                    if (rvalid && kb < K) { vb = arow[kb]; if (shift) vb = (vb + shift[kb]) * scale[kb]; }
                    af[s][kt].u[i]     = f2bf(va);
                    af[s][kt].u[8 + i] = f2bf(vb);
                }
            }
        }
    }

#pragma unroll
    for (int nt = 0; nt < 8; ++nt) {
        v8f acc0 = {}, acc1 = {};
#pragma unroll
        for (int kt = 0; kt < KT; ++kt) {
            Frag16 b;
            const u16x8* p = (const u16x8*)(sB + ((((kt << 3) + nt) << 9) + (lane << 4)));
            b.h2[0] = p[0];
            b.h2[1] = p[1];
            acc0 = __builtin_amdgcn_wmma_f32_16x16x32_bf16(false, af[0][kt].v, false, b.v,
                                                           (short)0, acc0, false, false);
            acc1 = __builtin_amdgcn_wmma_f32_16x16x32_bf16(false, af[1][kt].v, false, b.v,
                                                           (short)0, acc1, false, false);
        }
        const int col = (nt << 4) + (lane & 15);
        const float bv = bias ? bias[col] : 0.f;
#pragma unroll
        for (int r = 0; r < 8; ++r) {
            int r0 = rowBase + r + (half << 3);
            int r1 = r0 + 16;
            if (r0 < M) {
                float vv = acc0[r] + bv;
                if (relu) vv = fmaxf(vv, 0.f);
                out[(long long)r0 * 128 + col] = vv;
            }
            if (r1 < M) {
                float vv = acc1[r] + bv;
                if (relu) vv = fmaxf(vv, 0.f);
                out[(long long)r1 * 128 + col] = vv;
            }
        }
    }
}

// ---------------- fused edge kernel ----------------
// For each edge e: a = relu((matR[scat[e]] + matL[oth[e]]) * sf); msg = a @ wf + bf;
// atomically accumulate msg into sums[scat[e]].  4 waves/block, 16 edges/wave.
__global__ void __launch_bounds__(128)
edge_msg_kernel(const int* __restrict__ scat, const int* __restrict__ oth,
                const float* __restrict__ matR, const float* __restrict__ matL,
                const unsigned short* __restrict__ wf, const float* __restrict__ bf,
                const float* __restrict__ sfp, float* __restrict__ sums, int E) {
    __shared__ unsigned short sB[4 * 8 * 32 * 16];       // wf swizzled: 32 KB
    __shared__ unsigned short sA[4][2048];               // 16 edges x 128 k, per wave: 16 KB
    __shared__ int sDst[4][16];

    const int tid = threadIdx.x, lane = tid & 31, wave = tid >> 5;

    for (int i = tid; i < 128 * 128; i += 128) {
        int k = i >> 7, c = i & 127;
        sB[bswz(k, c)] = wf[i];
    }

    const float sf = *sfp;
    const int ebase = blockIdx.x * 64 + wave * 16;
    const bool fullwave = (ebase + 16) <= E;
    const int k4 = lane << 2;                            // 4 contiguous k per lane

    for (int m = 0; m < 16; ++m) {
        int e = ebase + m;
        bool valid = e < E;
        int is = 0, io = 0;
        if (valid) { is = scat[e]; io = oth[e]; }
        if (lane == 0) sDst[wave][m] = valid ? is : -1;
        float4 vr = make_float4(0.f, 0.f, 0.f, 0.f);
        float4 vl = make_float4(0.f, 0.f, 0.f, 0.f);
        if (valid) {
            vr = *(const float4*)(matR + (long long)is * 128 + k4);
            vl = *(const float4*)(matL + (long long)io * 128 + k4);
        }
        u16x4 pk;
        pk[0] = f2bf(fmaxf((vr.x + vl.x) * sf, 0.f));
        pk[1] = f2bf(fmaxf((vr.y + vl.y) * sf, 0.f));
        pk[2] = f2bf(fmaxf((vr.z + vl.z) * sf, 0.f));
        pk[3] = f2bf(fmaxf((vr.w + vl.w) * sf, 0.f));
        *(u16x4*)(&sA[wave][aswz(m, k4)]) = pk;          // 4 k's stay fragment-contiguous
    }
    __syncthreads();

    Frag16 af[4];
#pragma unroll
    for (int kt = 0; kt < 4; ++kt) {
        const u16x8* p = (const u16x8*)(&sA[wave][(kt << 9) + (lane << 4)]);
        af[kt].h2[0] = p[0];
        af[kt].h2[1] = p[1];
    }

    const int half = lane >> 4;
    int dstRow[8];                                       // hoisted: invariant across nt
#pragma unroll
    for (int r = 0; r < 8; ++r) dstRow[r] = sDst[wave][r + (half << 3)];

#pragma unroll
    for (int nt = 0; nt < 8; ++nt) {
        v8f acc = {};
#pragma unroll
        for (int kt = 0; kt < 4; ++kt) {
            Frag16 b;
            const u16x8* p = (const u16x8*)(sB + ((((kt << 3) + nt) << 9) + (lane << 4)));
            b.h2[0] = p[0];
            b.h2[1] = p[1];
            acc = __builtin_amdgcn_wmma_f32_16x16x32_bf16(false, af[kt].v, false, b.v,
                                                          (short)0, acc, false, false);
        }
        const int col = (nt << 4) + (lane & 15);
        const float bv = bf[col];
        if (fullwave) {                                  // wave-uniform: no per-lane exec games
#pragma unroll
            for (int r = 0; r < 8; ++r)
                atomicAdd(&sums[dstRow[r] * 128 + col], acc[r] + bv);
        } else {
#pragma unroll
            for (int r = 0; r < 8; ++r)
                if (dstRow[r] >= 0) atomicAdd(&sums[dstRow[r] * 128 + col], acc[r] + bv);
        }
    }
}

// ---------------- degree count ----------------
__global__ void count_kernel(const int* __restrict__ idx, float* __restrict__ cnt, int E) {
    int i = blockIdx.x * blockDim.x + threadIdx.x;
    if (i < E) atomicAdd(&cnt[idx[i]], 1.f);
}

// ---------------- mean + scale + concat: h = [sums/max(cnt,1)*sp , right] ----------------
__global__ void meanconcat_kernel(const float* __restrict__ sums, const float* __restrict__ cnt,
                                  const float* __restrict__ spp, const float* __restrict__ right,
                                  float* __restrict__ h, int NR) {
    int i = blockIdx.x * blockDim.x + threadIdx.x;
    if (i >= NR * 128) return;
    int n = i >> 7, c = i & 127;
    float sp = *spp;
    h[(n << 8) + c]       = sums[i] / fmaxf(cnt[n], 1.f) * sp;
    h[(n << 8) + 128 + c] = right[i];
}

// ---------------- head: sigmoid(t[M,128] @ w2[128,2] + b2) ----------------
__global__ void head_kernel(const float* __restrict__ t, const float* __restrict__ w2,
                            const float* __restrict__ b2, float* __restrict__ out, int M) {
    int n = blockIdx.x * blockDim.x + threadIdx.x;
    if (n >= M) return;
    float a0 = b2[0], a1 = b2[1];
    const float* row = t + (long long)n * 128;
#pragma unroll 4
    for (int k = 0; k < 128; ++k) {
        float v = row[k];
        a0 += v * w2[2 * k];
        a1 += v * w2[2 * k + 1];
    }
    out[2 * n]     = 1.f / (1.f + __expf(-a0));
    out[2 * n + 1] = 1.f / (1.f + __expf(-a1));
}

// ---------------- host-side dispatch ----------------
static inline void run_gemm(int KT, const float* A, const unsigned short* Bw, const float* bias,
                            const float* shift, const float* scale, float* out,
                            int M, int K, int relu, hipStream_t s) {
    dim3 g((M + 127) / 128), b(128);
    size_t lds = (size_t)KT * 32 * 128 * 2;
    switch (KT) {
        case 3: gemm128<3><<<g, b, lds, s>>>(A, Bw, bias, shift, scale, out, M, K, relu); break;
        case 4: gemm128<4><<<g, b, lds, s>>>(A, Bw, bias, shift, scale, out, M, K, relu); break;
        case 8: gemm128<8><<<g, b, lds, s>>>(A, Bw, bias, shift, scale, out, M, K, relu); break;
        default: break;
    }
}

static inline void run_cvt(const float* src, unsigned short* dst, int K, int Kpad, hipStream_t s) {
    int n = Kpad * 128;
    cvt_pad_bf16<<<(n + 255) / 256, 256, 0, s>>>(src, dst, K, Kpad);
}

extern "C" void kernel_launch(void* const* d_in, const int* in_sizes, int n_in,
                              void* d_out, int out_size, void* d_ws, size_t ws_size,
                              hipStream_t stream) {
    const int F  = in_sizes[3];          // 69 (pe_shift length)
    const int NP = in_sizes[0] / F;
    const int NC = in_sizes[1] / F;
    const int E  = in_sizes[2] / 2;
    const int FP = ((F + 31) / 32) * 32; // 96

    const float* pivotF = (const float*)d_in[0];
    const float* childF = (const float*)d_in[1];
    const int*   ei     = (const int*)d_in[2];
    const int*   srcI   = ei;
    const int*   dstI   = ei + E;

    const float* pe_shift = (const float*)d_in[3];
    const float* pe_scale = (const float*)d_in[4];
    const float* pe_w1    = (const float*)d_in[5];
    const float* pe_b1    = (const float*)d_in[6];
    const float* pe_w2    = (const float*)d_in[7];
    const float* pe_b2    = (const float*)d_in[8];
    const float* ce_shift = (const float*)d_in[9];
    const float* ce_scale = (const float*)d_in[10];
    const float* ce_w1    = (const float*)d_in[11];
    const float* ce_b1    = (const float*)d_in[12];
    const float* ce_w2    = (const float*)d_in[13];
    const float* ce_b2    = (const float*)d_in[14];

    const float* vc_wl = (const float*)d_in[15];
    const float* vc_bl = (const float*)d_in[16];
    const float* vc_wr = (const float*)d_in[17];
    const float* vc_sf = (const float*)d_in[18];
    const float* vc_wf = (const float*)d_in[19];
    const float* vc_bf = (const float*)d_in[20];
    const float* vc_sp = (const float*)d_in[21];
    const float* vc_wo1 = (const float*)d_in[22];
    const float* vc_bo1 = (const float*)d_in[23];
    const float* vc_wo2 = (const float*)d_in[24];
    const float* vc_bo2 = (const float*)d_in[25];

    const float* cv_wl = (const float*)d_in[26];
    const float* cv_bl = (const float*)d_in[27];
    const float* cv_wr = (const float*)d_in[28];
    const float* cv_sf = (const float*)d_in[29];
    const float* cv_wf = (const float*)d_in[30];
    const float* cv_bf = (const float*)d_in[31];
    const float* cv_sp = (const float*)d_in[32];
    const float* cv_wo1 = (const float*)d_in[33];
    const float* cv_bo1 = (const float*)d_in[34];
    const float* cv_wo2 = (const float*)d_in[35];
    const float* cv_bo2 = (const float*)d_in[36];

    const float* out_w1 = (const float*)d_in[37];
    const float* out_b1 = (const float*)d_in[38];
    const float* out_w2 = (const float*)d_in[39];
    const float* out_b2 = (const float*)d_in[40];

    // ---- workspace carve-up ----
    char* ws = (char*)d_ws;
    size_t off = 0;
    auto alloc = [&](size_t bytes) -> char* {
        char* p = ws + off;
        off = (off + bytes + 255) & ~(size_t)255;
        return p;
    };
    unsigned short* wPE1  = (unsigned short*)alloc((size_t)FP * 128 * 2);
    unsigned short* wPE2  = (unsigned short*)alloc((size_t)128 * 128 * 2);
    unsigned short* wCE1  = (unsigned short*)alloc((size_t)FP * 128 * 2);
    unsigned short* wCE2  = (unsigned short*)alloc((size_t)128 * 128 * 2);
    unsigned short* wVCl  = (unsigned short*)alloc((size_t)128 * 128 * 2);
    unsigned short* wVCr  = (unsigned short*)alloc((size_t)128 * 128 * 2);
    unsigned short* wVCf  = (unsigned short*)alloc((size_t)128 * 128 * 2);
    unsigned short* wVCo1 = (unsigned short*)alloc((size_t)256 * 128 * 2);
    unsigned short* wVCo2 = (unsigned short*)alloc((size_t)128 * 128 * 2);
    unsigned short* wCVl  = (unsigned short*)alloc((size_t)128 * 128 * 2);
    unsigned short* wCVr  = (unsigned short*)alloc((size_t)128 * 128 * 2);
    unsigned short* wCVf  = (unsigned short*)alloc((size_t)128 * 128 * 2);
    unsigned short* wCVo1 = (unsigned short*)alloc((size_t)256 * 128 * 2);
    unsigned short* wCVo2 = (unsigned short*)alloc((size_t)128 * 128 * 2);
    unsigned short* wOut1 = (unsigned short*)alloc((size_t)128 * 128 * 2);

    float* pv   = (float*)alloc((size_t)NP * 128 * 4);   // pivot hidden
    float* ch   = (float*)alloc((size_t)NC * 128 * 4);   // children hidden
    float* bufC = (float*)alloc((size_t)NC * 128 * 4);   // NC-sized scratch
    float* bufP = (float*)alloc((size_t)NP * 128 * 4);   // NP-sized scratch
    float* sums = (float*)alloc((size_t)NC * 128 * 4);   // scatter accum (>= NP*128)
    float* hbuf = (float*)alloc((size_t)NC * 256 * 4);   // concat buffer (>= NP*256)
    float* cnt  = (float*)alloc((size_t)NC * 4);         // degrees (>= NP)

    // ---- weight conversion (f32 -> zero-padded bf16) ----
    run_cvt(pe_w1, wPE1, F, FP, stream);
    run_cvt(pe_w2, wPE2, 128, 128, stream);
    run_cvt(ce_w1, wCE1, F, FP, stream);
    run_cvt(ce_w2, wCE2, 128, 128, stream);
    run_cvt(vc_wl, wVCl, 128, 128, stream);
    run_cvt(vc_wr, wVCr, 128, 128, stream);
    run_cvt(vc_wf, wVCf, 128, 128, stream);
    run_cvt(vc_wo1, wVCo1, 256, 256, stream);
    run_cvt(vc_wo2, wVCo2, 128, 128, stream);
    run_cvt(cv_wl, wCVl, 128, 128, stream);
    run_cvt(cv_wr, wCVr, 128, 128, stream);
    run_cvt(cv_wf, wCVf, 128, 128, stream);
    run_cvt(cv_wo1, wCVo1, 256, 256, stream);
    run_cvt(cv_wo2, wCVo2, 128, 128, stream);
    run_cvt(out_w1, wOut1, 128, 128, stream);

    // ---- embeddings (PreNorm fused into A-load) ----
    run_gemm(FP / 32, pivotF, wPE1, pe_b1, pe_shift, pe_scale, bufP, NP, F, 1, stream);
    run_gemm(4, bufP, wPE2, pe_b2, nullptr, nullptr, pv, NP, 128, 1, stream);
    run_gemm(FP / 32, childF, wCE1, ce_b1, ce_shift, ce_scale, bufC, NC, F, 1, stream);
    run_gemm(4, bufC, wCE2, ce_b2, nullptr, nullptr, ch, NC, 128, 1, stream);

    const int eg = (E + 63) / 64;

    // ---- conv v->c (aggregate to children over dst) ----
    run_gemm(4, ch, wVCl, vc_bl, nullptr, nullptr, bufC, NC, 128, 0, stream);    // chL = ch@wl+bl
    run_gemm(4, pv, wVCr, nullptr, nullptr, nullptr, bufP, NP, 128, 0, stream);  // pvR = pv@wr
    zero_f32<<<(NC * 128 + 255) / 256, 256, 0, stream>>>(sums, NC * 128);
    zero_f32<<<(NC + 255) / 256, 256, 0, stream>>>(cnt, NC);
    edge_msg_kernel<<<eg, 128, 0, stream>>>(dstI, srcI, bufC, bufP, wVCf, vc_bf, vc_sf, sums, E);
    count_kernel<<<(E + 255) / 256, 256, 0, stream>>>(dstI, cnt, E);
    meanconcat_kernel<<<(NC * 128 + 255) / 256, 256, 0, stream>>>(sums, cnt, vc_sp, ch, hbuf, NC);
    run_gemm(8, hbuf, wVCo1, vc_bo1, nullptr, nullptr, bufC, NC, 256, 1, stream);
    run_gemm(4, bufC, wVCo2, vc_bo2, nullptr, nullptr, ch, NC, 128, 0, stream);  // ch updated

    // ---- conv c->v (reversed edges; aggregate to pivots over src) ----
    run_gemm(4, pv, wCVl, cv_bl, nullptr, nullptr, bufP, NP, 128, 0, stream);    // pvL = pv@wl+bl
    run_gemm(4, ch, wCVr, nullptr, nullptr, nullptr, bufC, NC, 128, 0, stream);  // chR = ch@wr
    zero_f32<<<(NP * 128 + 255) / 256, 256, 0, stream>>>(sums, NP * 128);
    zero_f32<<<(NP + 255) / 256, 256, 0, stream>>>(cnt, NP);
    edge_msg_kernel<<<eg, 128, 0, stream>>>(srcI, dstI, bufP, bufC, wCVf, cv_bf, cv_sf, sums, E);
    count_kernel<<<(E + 255) / 256, 256, 0, stream>>>(srcI, cnt, E);
    meanconcat_kernel<<<(NP * 128 + 255) / 256, 256, 0, stream>>>(sums, cnt, cv_sp, pv, hbuf, NP);
    run_gemm(8, hbuf, wCVo1, cv_bo1, nullptr, nullptr, bufP, NP, 256, 1, stream);
    run_gemm(4, bufP, wCVo2, cv_bo2, nullptr, nullptr, pv, NP, 128, 0, stream);  // pv updated

    // ---- output head ----
    run_gemm(4, pv, wOut1, out_b1, nullptr, nullptr, bufP, NP, 128, 1, stream);
    head_kernel<<<(NP + 255) / 256, 256, 0, stream>>>(bufP, out_w2, out_b2, (float*)d_out, NP);
}
